// LstmCell_normal_979252543670
// MI455X (gfx1250) — matrix operand order repaired and checked by reading, not compiled, not run on hardware
//
#include <hip/hip_runtime.h>

// ---------------------------------------------------------------------------
// LSTM cell, MI455X (gfx1250, wave32, WMMA, async-to-LDS staging).
//   z = [x|h] @ [Wx;Wh] + b   -> one bf16 GEMM [4096x4096]@[4096x8192], f32 acc
//   gates fused in epilogue (sigmoid/tanh via v_exp_f32)
// Workspace layout (needs 96 MB):
//   [0, 32MB)   : A  = bf16 [4096][4096]  (x cols 0..2047, h cols 2048..4095)
//   [32, 96MB)  : Wt = bf16 [8192][4096]  (N-major: Wt[n][k], n = gate*2048+u)
// ---------------------------------------------------------------------------

typedef __attribute__((ext_vector_type(16))) __bf16 v16bf;
typedef __attribute__((ext_vector_type(8)))  __bf16 v8bf;
typedef __attribute__((ext_vector_type(8)))  float  v8f;

#define B_DIM 4096
#define D_DIM 2048
#define U_DIM 2048
#define K_DIM 4096   // D + U
#define N4U   8192   // 4 * U

#define KT    32     // K tile per pipeline stage
#define AST   40     // LDS row stride (elements): 80 B = 16B-aligned, 20-bank step
#define BST   40

union BF16x16 { v16bf v; v8bf h[2]; };

__device__ __forceinline__ float sigmoid_f(float x) {
    return 1.0f / (1.0f + __expf(-x));
}
__device__ __forceinline__ float tanh_f(float x) {
    float e = __expf(2.0f * x);
    return (e - 1.0f) / (e + 1.0f);
}

// CDNA5 async global->LDS copy, 16B per lane, tracked by ASYNCcnt.
// VDST = LDS byte address (low 32 bits of generic pointer), VADDR = 64-bit global.
#define ASYNC_B128(ldsptr, gptr)                                        \
    asm volatile("global_load_async_to_lds_b128 %0, %1, off"            \
                 :: "v"((unsigned)(uintptr_t)(ldsptr)), "v"(gptr)       \
                 : "memory")

#define WAIT_ASYNC0() asm volatile("s_wait_asynccnt 0" ::: "memory")

// ---------------------------------------------------------------------------
// Kernel 1: pack [x | h] -> bf16 A[4096][4096]
// ---------------------------------------------------------------------------
__global__ __launch_bounds__(256) void pack_A(const float* __restrict__ x,
                                              const float* __restrict__ h,
                                              __bf16* __restrict__ A) {
    size_t idx = (size_t)blockIdx.x * 256 + threadIdx.x;   // 0 .. 4096*4096-1
    int r = (int)(idx >> 12);
    int c = (int)(idx & 4095);
    float v = (c < D_DIM) ? x[(size_t)r * D_DIM + c]
                          : h[(size_t)r * U_DIM + (c - D_DIM)];
    A[idx] = (__bf16)v;
}

// ---------------------------------------------------------------------------
// Kernel 2: transpose-pack weights -> bf16 Wt[8192][4096]
//   Wt[n][k]: n = gate*2048 + u ; k<2048 -> Wx_gate[k][u], else Wh_gate[k-2048][u]
// 32x32 LDS tile transpose, block (32,8).
// ---------------------------------------------------------------------------
__global__ __launch_bounds__(256) void pack_Wt(const float* __restrict__ Wxf,
                                               const float* __restrict__ Wxi,
                                               const float* __restrict__ Wxo,
                                               const float* __restrict__ Wxg,
                                               const float* __restrict__ Whf,
                                               const float* __restrict__ Whi,
                                               const float* __restrict__ Who,
                                               const float* __restrict__ Whg,
                                               __bf16* __restrict__ Wt) {
    __shared__ __bf16 tile[32][33];

    int k0 = blockIdx.x * 32;                 // K tile origin (0..4095)
    int n0 = blockIdx.y * 32;                 // N tile origin (0..8191)
    int tx = threadIdx.x;                     // 0..31
    int ty = threadIdx.y;                     // 0..7

    int gate   = n0 >> 11;                    // 0..3 (tile never crosses a gate)
    int uBase  = n0 & 2047;
    bool fromX = (k0 < D_DIM);                // tile never crosses x/h boundary

    const float* Wx[4] = {Wxf, Wxi, Wxo, Wxg};
    const float* Wh[4] = {Whf, Whi, Who, Whg};
    const float* W = fromX ? Wx[gate] : Wh[gate];
    int kSrc0 = fromX ? k0 : (k0 - D_DIM);

    #pragma unroll
    for (int i = 0; i < 4; ++i) {
        int kl = ty + i * 8;                  // local k
        tile[kl][tx] = (__bf16)W[(size_t)(kSrc0 + kl) * U_DIM + uBase + tx];
    }
    __syncthreads();
    #pragma unroll
    for (int i = 0; i < 4; ++i) {
        int nl = ty + i * 8;                  // local n
        Wt[(size_t)(n0 + nl) * K_DIM + k0 + tx] = tile[tx][nl];
    }
}

// ---------------------------------------------------------------------------
// Kernel 3: fused GEMM + gates, double-buffered async-to-LDS pipeline.
// Block: 256 threads = 8 waves, wave grid 4(M) x 2(N).
// Wave tile: 32 rows x 32 U-cols, 4 gates -> 16 accumulators (v8f).
// Block tile: 128 rows x 64 U-cols (per gate). Grid: (U/64, B/128) = (32, 32).
// LDS/stage: A 128x32 bf16 (padded), B 4x64x32 bf16 (padded); 2 stages = 60 KB.
// Per stage: all 10 fragments preloaded (ds_load_b128 x20), then 16 WMMAs.
// ---------------------------------------------------------------------------
__global__ __launch_bounds__(256) void lstm_gemm(const __bf16* __restrict__ A,
                                                 const __bf16* __restrict__ Wt,
                                                 const float* __restrict__ cin,
                                                 const float* __restrict__ bf_,
                                                 const float* __restrict__ bi_,
                                                 const float* __restrict__ bo_,
                                                 const float* __restrict__ bg_,
                                                 float* __restrict__ hout,
                                                 float* __restrict__ cout) {
    __shared__ __bf16 As[2][128 * AST];
    __shared__ __bf16 Bs[2][256 * BST];

    const int tid   = threadIdx.x;
    const int lane  = tid & 31;
    const int wave  = tid >> 5;       // 0..7
    const int waveM = wave >> 1;      // 0..3
    const int waveN = wave & 1;       // 0..1
    const int l16   = lane & 15;
    const int half  = lane >> 4;      // 0 or 1

    const int mBlock = blockIdx.y * 128;                // block row origin
    const int nBlock = blockIdx.x * 64;                 // block U-col origin
    const int mWave  = waveM * 32;                      // wave row offset in block
    const int nWave  = waveN * 32;                      // wave U-col offset in block

    // copy mapping: 256 threads, 16 B each => 64 rows (of 32 bf16) per issue
    const int cpRow = tid >> 2;       // 0..63
    const int cpSeg = tid & 3;        // 0..3 (16B segment within 64B row)

    // stage issue: 6 async b128 ops per thread
    auto issue_stage = [&](int buf, int k0) {
        #pragma unroll
        for (int i = 0; i < 2; ++i) {                    // A: 128 rows
            int r = cpRow + i * 64;
            const __bf16* g = A + (size_t)(mBlock + r) * K_DIM + k0 + cpSeg * 8;
            ASYNC_B128(&As[buf][r * AST + cpSeg * 8], g);
        }
        #pragma unroll
        for (int gg = 0; gg < 4; ++gg) {                 // B: 4 gates x 64 rows
            const __bf16* g = Wt + (size_t)(gg * U_DIM + nBlock + cpRow) * K_DIM
                                 + k0 + cpSeg * 8;
            ASYNC_B128(&Bs[buf][(gg * 64 + cpRow) * BST + cpSeg * 8], g);
        }
    };

    v8f acc[2][2][4];
    #pragma unroll
    for (int mt = 0; mt < 2; ++mt)
        #pragma unroll
        for (int nt = 0; nt < 2; ++nt)
            #pragma unroll
            for (int g = 0; g < 4; ++g)
                acc[mt][nt][g] = (v8f){0.f,0.f,0.f,0.f,0.f,0.f,0.f,0.f};

    issue_stage(0, 0);

    int s = 0;
    for (int k0 = 0; k0 < K_DIM; k0 += KT, s ^= 1) {
        WAIT_ASYNC0();          // our stage-s ops done (in-order completion)
        __syncthreads();        // all waves' stage-s data visible; also proves
                                // every wave finished reading buffer s^1

        if (k0 + KT < K_DIM)    // overlap next stage's DMA with this compute
            issue_stage(s ^ 1, k0 + KT);

        // Preload ALL fragments of this k-step into distinct registers so the
        // scheduler can grade s_wait_dscnt instead of stalling per WMMA pair.
        // A frag (16-bit A layout): lane<16 -> K {0..7,16..23},
        //                           lane>=16 -> K {8..15,24..31}
        BF16x16 a[2];
        #pragma unroll
        for (int mt = 0; mt < 2; ++mt) {
            const __bf16* p = &As[s][(mWave + mt * 16 + l16) * AST + half * 8];
            a[mt].h[0] = *(const v8bf*)(p);
            a[mt].h[1] = *(const v8bf*)(p + 16);
        }
        // B frag: lane<16 -> K {0..7,16..23}, lane>=16 -> K {8..15,24..31}, col = l16
        BF16x16 b[4][2];
        #pragma unroll
        for (int gg = 0; gg < 4; ++gg) {
            #pragma unroll
            for (int nt = 0; nt < 2; ++nt) {
                const __bf16* p = &Bs[s][(gg * 64 + nWave + nt * 16 + l16) * BST
                                         + half * 8];
                b[gg][nt].h[0] = *(const v8bf*)(p);
                b[gg][nt].h[1] = *(const v8bf*)(p + 16);
            }
        }
        // 16-WMMA burst
        #pragma unroll
        for (int gg = 0; gg < 4; ++gg)
            #pragma unroll
            for (int nt = 0; nt < 2; ++nt)
                #pragma unroll
                for (int mt = 0; mt < 2; ++mt)
                    acc[mt][nt][gg] = __builtin_amdgcn_wmma_f32_16x16x32_bf16(
                        false, a[mt].v, false, b[gg][nt].v,
                        (short)0, acc[mt][nt][gg], false, false);
        // single barrier per stage: next iteration's wait+barrier precedes any
        // overwrite of the buffer we just read.
    }

    // Epilogue: bias + gates + state update.
    // C/D layout: VGPR r, lane -> row = r + (lane>=16 ? 8 : 0), col = lane&15.
    #pragma unroll
    for (int nt = 0; nt < 2; ++nt) {
        const int col = nBlock + nWave + nt * 16 + l16;          // 0..2047
        const float bfv = bf_[col];
        const float biv = bi_[col];
        const float bov = bo_[col];
        const float bgv = bg_[col];
        #pragma unroll
        for (int mt = 0; mt < 2; ++mt) {
            #pragma unroll
            for (int r = 0; r < 8; ++r) {
                const int row = mBlock + mWave + mt * 16 + r + half * 8;
                const float zf = acc[mt][nt][0][r] + bfv;
                const float zi = acc[mt][nt][1][r] + biv;
                const float zo = acc[mt][nt][2][r] + bov;
                const float zg = acc[mt][nt][3][r] + bgv;
                const float fg = sigmoid_f(zf);
                const float ig = sigmoid_f(zi);
                const float og = sigmoid_f(zo);
                const float gg = tanh_f(zg);
                const size_t off = (size_t)row * U_DIM + col;
                const float cn = fg * cin[off] + ig * gg;
                const float hn = og * tanh_f(cn);
                hout[off] = hn;
                cout[off] = cn;
            }
        }
    }
}

// ---------------------------------------------------------------------------
extern "C" void kernel_launch(void* const* d_in, const int* in_sizes, int n_in,
                              void* d_out, int out_size, void* d_ws, size_t ws_size,
                              hipStream_t stream) {
    const float* x   = (const float*)d_in[0];
    const float* h   = (const float*)d_in[1];
    const float* c   = (const float*)d_in[2];
    const float* Wxf = (const float*)d_in[3];
    const float* Wxi = (const float*)d_in[4];
    const float* Wxo = (const float*)d_in[5];
    const float* Wxg = (const float*)d_in[6];
    const float* bf  = (const float*)d_in[7];
    const float* bi  = (const float*)d_in[8];
    const float* bo  = (const float*)d_in[9];
    const float* bg  = (const float*)d_in[10];
    const float* Whf = (const float*)d_in[11];
    const float* Whi = (const float*)d_in[12];
    const float* Who = (const float*)d_in[13];
    const float* Whg = (const float*)d_in[14];

    __bf16* A  = (__bf16*)d_ws;
    __bf16* Wt = (__bf16*)((char*)d_ws + (size_t)B_DIM * K_DIM * sizeof(__bf16)); // +32MB

    float* hout = (float*)d_out;
    float* cout = hout + (size_t)B_DIM * U_DIM;

    // 1) pack [x|h] -> bf16 A
    pack_A<<<(B_DIM * (size_t)K_DIM) / 256, 256, 0, stream>>>(x, h, A);

    // 2) transpose-pack weights -> bf16 Wt[n][k]
    pack_Wt<<<dim3(K_DIM / 32, N4U / 32), dim3(32, 8), 0, stream>>>(
        Wxf, Wxi, Wxo, Wxg, Whf, Whi, Who, Whg, Wt);

    // 3) fused GEMM + gate epilogue (async-to-LDS pipelined WMMA)
    lstm_gemm<<<dim3(U_DIM / 64, B_DIM / 128), 256, 0, stream>>>(
        A, Wt, c, bf, bi, bo, bg, hout, cout);
}
